// GraphClassifier_70806830842510
// MI455X (gfx1250) — matrix operand order repaired and checked by reading, not compiled, not run on hardware
//
#include <hip/hip_runtime.h>

typedef __attribute__((ext_vector_type(16))) _Float16 v16h;
typedef __attribute__((ext_vector_type(8)))  _Float16 v8h;
typedef __attribute__((ext_vector_type(8)))  float    v8f;
typedef __attribute__((ext_vector_type(4)))  int      v4i;

#define WMMA_F16(a, b, c) \
  __builtin_amdgcn_wmma_f32_16x16x32_f16(false, (a), false, (b), (short)0, (c), false, false)

// A fragment (16x32 f16) from an f16 row: lane half=0 holds K {0..7, 16..23},
// half=1 holds {8..15, 24..31}. Both chunks are 16B-aligned -> two b128 loads.
__device__ __forceinline__ v16h load_A_f16(const _Float16* __restrict__ row, int half) {
  v8h lo = *(const v8h*)(row + 8 * half);
  v8h hi = *(const v8h*)(row + 16 + 8 * half);
  v16h a;
#pragma unroll
  for (int i = 0; i < 8; ++i) { a[i] = lo[i]; a[8 + i] = hi[i]; }
  return a;
}

// B fragment (32K x 16N f16) packed from a row-major f32 [K x ldb] matrix:
// lanes 0-15 hold K={0..7,16..23} of column col, lanes 16-31 hold K={8..15,24..31}.
__device__ __forceinline__ v16h pack_B_frag(const float* __restrict__ B, int ldb,
                                            int col, int half) {
  v16h b;
#pragma unroll
  for (int i = 0; i < 16; ++i) b[i] = (_Float16)B[(8 * half + i + (i >= 8 ? 8 : 0)) * ldb + col];
  return b;
}

__global__ void build_x0_kernel(const float* __restrict__ feat,
                                const float* __restrict__ embed,
                                const int* __restrict__ idx,
                                _Float16* __restrict__ x0, int N) {
  int i = blockIdx.x * blockDim.x + threadIdx.x;
  if (i >= N * 32) return;
  int n = i >> 5, j = i & 31;
  float v = (j < 16) ? feat[n * 16 + j] : embed[(unsigned)idx[n] * 16u + (j - 16)];
  x0[i] = (_Float16)v;
}

__global__ void f32_to_f16_kernel(const float* __restrict__ in,
                                  _Float16* __restrict__ out, int n) {
  int i = blockIdx.x * blockDim.x + threadIdx.x;
  if (i < n) out[i] = (_Float16)in[i];
}

__global__ void __launch_bounds__(256)
rgcn_edge_kernel(const _Float16* __restrict__ x,        // [N,32] f16
                 const int* __restrict__ src,
                 const int* __restrict__ dst,
                 const int* __restrict__ etype,
                 const _Float16* __restrict__ attn_tab, // [R,32] f16
                 const float* __restrict__ bases,       // [4,32,32]
                 const float* __restrict__ wcomp,       // [R,4]
                 const float* __restrict__ Aw,          // [96,32]
                 const float* __restrict__ Ab,          // [32]
                 const float* __restrict__ Bw,          // [32]
                 const float* __restrict__ Bb,          // [1]
                 float* __restrict__ nei,               // [N,32] accumulated
                 int ntiles) {
  const int lane = threadIdx.x & 31;
  const int l16  = lane & 15;
  const int half = lane >> 4;
  const int wave   = (blockIdx.x * blockDim.x + threadIdx.x) >> 5;
  const int nwaves = (gridDim.x * blockDim.x) >> 5;

  // Loop-invariant B fragments, kept in VGPRs for the whole tile loop.
  v16h BAw[3][2];
#pragma unroll
  for (int kc = 0; kc < 3; ++kc)
#pragma unroll
    for (int nh = 0; nh < 2; ++nh)
      BAw[kc][nh] = pack_B_frag(Aw + kc * 32 * 32, 32, nh * 16 + l16, half);

  v16h Bbase[4][2];
#pragma unroll
  for (int b = 0; b < 4; ++b)
#pragma unroll
    for (int nh = 0; nh < 2; ++nh)
      Bbase[b][nh] = pack_B_frag(bases + b * 32 * 32, 32, nh * 16 + l16, half);

  const float ab0 = Ab[l16], ab1 = Ab[16 + l16];
  const float bw0 = Bw[l16], bw1 = Bw[16 + l16];
  const float bb  = Bb[0];

  for (int t = wave; t < ntiles; t += nwaves) {
    const unsigned e = (unsigned)t * 16u + l16;  // both halves cover the same 16 edges
    const unsigned s = (unsigned)src[e];
    const unsigned d = (unsigned)dst[e];
    const unsigned r = (unsigned)etype[e];

    // per-row dst/etype for rows M = v + 8*half: contiguous vector loads (16B aligned)
    const int rbase = t * 16 + 8 * half;
    v4i d_lo = *(const v4i*)(dst + rbase);
    v4i d_hi = *(const v4i*)(dst + rbase + 4);
    v4i r_lo = *(const v4i*)(etype + rbase);
    v4i r_hi = *(const v4i*)(etype + rbase + 4);

    v16h A_xs = load_A_f16(x + (size_t)s * 32u, half);
    v16h A_xd = load_A_f16(x + (size_t)d * 32u, half);
    v16h A_at = load_A_f16(attn_tab + (size_t)r * 32u, half);

    // attention hidden: h = e_in @ Aw  (16x96 @ 96x32), K split over {xs, xd, attn}
    v8f c0 = {}, c1 = {};
    c0 = WMMA_F16(A_xs, BAw[0][0], c0);
    c1 = WMMA_F16(A_xs, BAw[0][1], c1);
    c0 = WMMA_F16(A_xd, BAw[1][0], c0);
    c1 = WMMA_F16(A_xd, BAw[1][1], c1);
    c0 = WMMA_F16(A_at, BAw[2][0], c0);
    c1 = WMMA_F16(A_at, BAw[2][1], c1);

    // a = sigmoid(relu(h + Ab) @ Bw + Bb): row dot via 16-lane xor reduction.
    float a_row[8];
#pragma unroll
    for (int v = 0; v < 8; ++v) {
      float h0 = fmaxf(c0[v] + ab0, 0.f);
      float h1 = fmaxf(c1[v] + ab1, 0.f);
      float p = h0 * bw0 + h1 * bw1;
      p += __shfl_xor(p, 1);
      p += __shfl_xor(p, 2);
      p += __shfl_xor(p, 4);
      p += __shfl_xor(p, 8);
      a_row[v] = 1.f / (1.f + __expf(-(p + bb)));
    }

    // msg = sum_b wcomp[etype,b] * (xs @ bases_b) — shared-A WMMA per basis
    v8f m0 = {}, m1 = {};
#pragma unroll
    for (int b = 0; b < 4; ++b) {
      v8f z0 = {}, z1 = {};
      v8f y0 = WMMA_F16(A_xs, Bbase[b][0], z0);
      v8f y1 = WMMA_F16(A_xs, Bbase[b][1], z1);
#pragma unroll
      for (int v = 0; v < 8; ++v) {
        unsigned rr = (unsigned)(v < 4 ? r_lo[v & 3] : r_hi[v & 3]);
        float wc = wcomp[rr * 4u + b];
        m0[v] += wc * y0[v];
        m1[v] += wc * y1[v];
      }
    }

    // scatter a * msg into nei[dst]
#pragma unroll
    for (int v = 0; v < 8; ++v) {
      unsigned dd = (unsigned)(v < 4 ? d_lo[v & 3] : d_hi[v & 3]);
      float w0 = a_row[v] * m0[v];
      float w1 = a_row[v] * m1[v];
      atomicAdd(&nei[dd * 32u + l16], w0);
      atomicAdd(&nei[dd * 32u + 16u + l16], w1);
    }
  }
}

// OUT_F16=true  -> write f16 activations (input of next layer)
// OUT_F16=false -> write f32 (final output)
template <bool OUT_F16>
__global__ void __launch_bounds__(256)
rgcn_combine_kernel(const _Float16* __restrict__ x,  // [N,32] f16
                    const float* __restrict__ nei,   // [N,32]
                    const float* __restrict__ selfw, // [32,32]
                    void* __restrict__ out_raw,      // [N,32]
                    int ntiles) {
  const int lane = threadIdx.x & 31;
  const int l16  = lane & 15;
  const int half = lane >> 4;
  const int wave   = (blockIdx.x * blockDim.x + threadIdx.x) >> 5;
  const int nwaves = (gridDim.x * blockDim.x) >> 5;

  v16h Bs0 = pack_B_frag(selfw, 32, l16, half);
  v16h Bs1 = pack_B_frag(selfw, 32, 16 + l16, half);

  for (int t = wave; t < ntiles; t += nwaves) {
    const unsigned node = (unsigned)t * 16u + l16;
    v16h A = load_A_f16(x + (size_t)node * 32u, half);
    v8f c0 = {}, c1 = {};
    c0 = WMMA_F16(A, Bs0, c0);
    c1 = WMMA_F16(A, Bs1, c1);
#pragma unroll
    for (int v = 0; v < 8; ++v) {
      const unsigned nrow = (unsigned)t * 16u + v + 8u * half;
      float o0 = fmaxf(c0[v] + nei[nrow * 32u + l16], 0.f);
      float o1 = fmaxf(c1[v] + nei[nrow * 32u + 16u + l16], 0.f);
      if (OUT_F16) {
        _Float16* out = (_Float16*)out_raw;
        out[nrow * 32u + l16]       = (_Float16)o0;
        out[nrow * 32u + 16u + l16] = (_Float16)o1;
      } else {
        float* out = (float*)out_raw;
        out[nrow * 32u + l16]       = o0;
        out[nrow * 32u + 16u + l16] = o1;
      }
    }
  }
}

extern "C" void kernel_launch(void* const* d_in, const int* in_sizes, int n_in,
                              void* d_out, int out_size, void* d_ws, size_t ws_size,
                              hipStream_t stream) {
  const float* feat     = (const float*)d_in[0];
  const float* embed    = (const float*)d_in[1];
  const float* attn_tab = (const float*)d_in[2];
  const int*   idx      = (const int*)d_in[3];
  const int*   src      = (const int*)d_in[4];
  const int*   dst      = (const int*)d_in[5];
  const int*   etype    = (const int*)d_in[6];
  const float* bases0 = (const float*)d_in[7];
  const float* wcomp0 = (const float*)d_in[8];
  const float* selfw0 = (const float*)d_in[9];
  const float* Aw0    = (const float*)d_in[10];
  const float* Ab0    = (const float*)d_in[11];
  const float* Bw0    = (const float*)d_in[12];
  const float* Bb0    = (const float*)d_in[13];
  const float* bases1 = (const float*)d_in[14];
  const float* wcomp1 = (const float*)d_in[15];
  const float* selfw1 = (const float*)d_in[16];
  const float* Aw1    = (const float*)d_in[17];
  const float* Ab1    = (const float*)d_in[18];
  const float* Bw1    = (const float*)d_in[19];
  const float* Bb1    = (const float*)d_in[20];

  const int N = in_sizes[0] / 16;   // feat is [N,16]
  const int E = in_sizes[4];        // src is [E]
  const int R = in_sizes[2] / 32;   // attn_tab is [R,32]
  const int etiles = E / 16;        // E = 800000 -> exact
  const int ntiles = N / 16;        // N = 50000  -> exact

  // workspace layout: f32 nei accumulator, then f16 activations + attn table
  float*    nei   = (float*)d_ws;                        // [N,32] f32
  _Float16* xbuf  = (_Float16*)(nei + (size_t)N * 32);   // [N,32] f16
  _Float16* hbuf  = xbuf + (size_t)N * 32;               // [N,32] f16
  _Float16* attnh = hbuf + (size_t)N * 32;               // [R,32] f16

  build_x0_kernel<<<(N * 32 + 255) / 256, 256, 0, stream>>>(feat, embed, idx, xbuf, N);
  f32_to_f16_kernel<<<(R * 32 + 255) / 256, 256, 0, stream>>>(attn_tab, attnh, R * 32);

  // layer 0
  hipMemsetAsync(nei, 0, (size_t)N * 32 * sizeof(float), stream);
  rgcn_edge_kernel<<<512, 256, 0, stream>>>(xbuf, src, dst, etype, attnh,
                                            bases0, wcomp0, Aw0, Ab0, Bw0, Bb0,
                                            nei, etiles);
  rgcn_combine_kernel<true><<<256, 256, 0, stream>>>(xbuf, nei, selfw0, hbuf, ntiles);

  // layer 1
  hipMemsetAsync(nei, 0, (size_t)N * 32 * sizeof(float), stream);
  rgcn_edge_kernel<<<512, 256, 0, stream>>>(hbuf, src, dst, etype, attnh,
                                            bases1, wcomp1, Aw1, Ab1, Bw1, Bb1,
                                            nei, etiles);
  rgcn_combine_kernel<false><<<256, 256, 0, stream>>>(hbuf, nei, selfw1, d_out, ntiles);
}
